// Router_26817775796684
// MI455X (gfx1250) — matrix operand order repaired and checked by reading, not compiled, not run on hardware
//
#include <hip/hip_runtime.h>
#include <math.h>

typedef __attribute__((ext_vector_type(16))) _Float16 v16h;
typedef __attribute__((ext_vector_type(8)))  _Float16 v8h;
typedef __attribute__((ext_vector_type(8)))  float    v8f;

#define CIN   68
#define CIP   72            // ci padded to multiple of 8
#define CPROJ 128
#define HW    64            // 8x8 positions
#define NEXP  8
#define KTOT  672           // 9 kernel taps * CIP, padded region reads zeros
#define KSTEPS 21           // 672 / 32
#define ROWS  110           // 10x10 zero-padded grid + zero tail rows (kp=9 chunks)

// K mapping: K = kp*CIP + ci   (kp = dh*3+dw).  Chunk c (8 halfs) -> LDS half
// offset relative to s_x[pbase][0]:  (dh*10+dw)*CIP + ci,  all compile-time
// foldable once the ks loop is unrolled.
#define CHOFF(c) (((((c) / 9) / 3) * 10 + (((c) / 9) % 3)) * CIP + (((c) % 9) * 8))

// ---------------------------------------------------------------------------
// Pre-pass: conv_w [128][68][3][3] fp32 -> wB [co][KTOT] f16, K = kp*72 + ci.
// A B-fragment (lane: N=co, 16 contiguous K halfs) is one contiguous 32B load.
// ---------------------------------------------------------------------------
__global__ void convert_weights_kernel(const float* __restrict__ conv_w,
                                       _Float16* __restrict__ wB) {
    int idx = blockIdx.x * blockDim.x + threadIdx.x;
    if (idx >= CPROJ * KTOT) return;
    int K  = idx % KTOT;
    int co = idx / KTOT;
    int kp = K / CIP;
    int ci = K % CIP;
    float v = 0.0f;
    if (kp < 9 && ci < CIN)
        v = conv_w[co * (CIN * 9) + ci * 9 + kp];
    wB[idx] = (_Float16)v;
}

// ---------------------------------------------------------------------------
// Main fused kernel: one workgroup (256 thr = 8 waves) per image.
// conv (implicit GEMM, v_wmma_f32_16x16x32_f16, no materialized im2col)
// + bias + relu + mean-pool + fc + softmax + adaptive threshold + renorm.
// ---------------------------------------------------------------------------
__global__ void __launch_bounds__(256)
router_gate_kernel(const float* __restrict__ patch,
                   const _Float16* __restrict__ wB,
                   const float* __restrict__ conv_b,
                   const float* __restrict__ fc_w,
                   const float* __restrict__ fc_b,
                   const float* __restrict__ p_wmax,
                   const float* __restrict__ p_went,
                   const float* __restrict__ p_wgap,
                   const float* __restrict__ p_thr,
                   const int*   __restrict__ p_epoch,
                   float* __restrict__ out) {
    __shared__ alignas(16) _Float16 s_x[ROWS][CIP];   // 15840 B padded image
    __shared__ float s_pooled[CPROJ];
    __shared__ float s_logits[NEXP];

    const int tid  = threadIdx.x;
    const int lane = tid & 31;
    const int wave = tid >> 5;          // 0..7 -> N-tile (16 output channels)
    const int n    = blockIdx.x;

    const float* img = patch + (size_t)n * CIN * HW;
    const int coBase  = wave * 16;
    const int mrow    = lane & 15;
    const int halfSel = lane >> 4;      // 0: lanes 0-15, 1: lanes 16-31

    // ---- zero the padded grid, then scatter the image transposed into it ----
    {
        uint32_t* z = (uint32_t*)&s_x[0][0];
        for (int i = tid; i < ROWS * CIP / 2; i += 256) z[i] = 0u;
    }
    __syncthreads();
    for (int i = tid; i < CIN * HW; i += 256) {       // coalesced global read
        int ci  = i >> 6;
        int pos = i & 63;
        int hi  = pos >> 3, wi = pos & 7;
        s_x[(hi + 1) * 10 + (wi + 1)][ci] = (_Float16)img[i];
    }
    __syncthreads();

    // ---- implicit-GEMM: C[64x128] = A[64xK] * B[Kx128], K = 672 ----
    v8f acc0 = 0.0f, acc1 = 0.0f, acc2 = 0.0f, acc3 = 0.0f;
    const _Float16* sx = &s_x[0][0];
    int pbase[4];
    #pragma unroll
    for (int t = 0; t < 4; ++t) {
        int m = t * 16 + mrow;                        // output position 0..63
        pbase[t] = ((m >> 3) * 10 + (m & 7)) * CIP;   // (h*10+w)*CIP, border +1
    }                                                 // folded into tap offsets
    const _Float16* wrow = wB + (size_t)(coBase + mrow) * KTOT;

    #pragma unroll
    for (int ks = 0; ks < KSTEPS; ++ks) {
        // B fragment: lane = col N, 16 contiguous K halfs (2x global_load_b128)
        v16h bfrag = *(const v16h*)(wrow + ks * 32 + halfSel * 16);
        // A chunk offsets: compile-time constants selected by halfSel
        const int offA = halfSel ? CHOFF(ks * 4 + 2) : CHOFF(ks * 4);
        const int offB = halfSel ? CHOFF(ks * 4 + 3) : CHOFF(ks * 4 + 1);
        #pragma unroll
        for (int t = 0; t < 4; ++t) {
            v8h lo = *(const v8h*)(sx + pbase[t] + offA);   // ds_load_b128
            v8h hi = *(const v8h*)(sx + pbase[t] + offB);   // ds_load_b128
            v16h afrag;
            #pragma unroll
            for (int i = 0; i < 8; ++i) { afrag[i] = lo[i]; afrag[8 + i] = hi[i]; }
            v8f c = (t == 0) ? acc0 : (t == 1) ? acc1 : (t == 2) ? acc2 : acc3;
            c = __builtin_amdgcn_wmma_f32_16x16x32_f16(
                    false, afrag, false, bfrag, (short)0, c, false, false);
            if (t == 0) acc0 = c; else if (t == 1) acc1 = c;
            else if (t == 2) acc2 = c; else acc3 = c;
        }
    }

    // ---- bias + relu + mean-pool (conv output never touches memory) ----
    {
        const float bias = conv_b[coBase + mrow];
        float s = 0.0f;
        #pragma unroll
        for (int r = 0; r < 8; ++r) {
            s += fmaxf(acc0[r] + bias, 0.0f);
            s += fmaxf(acc1[r] + bias, 0.0f);
            s += fmaxf(acc2[r] + bias, 0.0f);
            s += fmaxf(acc3[r] + bias, 0.0f);
        }
        s += __shfl_xor(s, 16, 32);     // wave32: lane L and L^16 share column N
        if (lane < 16) s_pooled[coBase + mrow] = s * (1.0f / 64.0f);
    }
    __syncthreads();

    // ---- FC to 8 logits: 32 lanes, 4 lanes per expert ----
    if (tid < 32) {
        int e = tid >> 2, g = tid & 3;
        float p = 0.0f;
        for (int c = g; c < CPROJ; c += 4)
            p += s_pooled[c] * fc_w[c * NEXP + e];
        p += __shfl_xor(p, 1, 32);
        p += __shfl_xor(p, 2, 32);
        if (g == 0) s_logits[e] = p + fc_b[e];
    }
    __syncthreads();

    // ---- softmax + adaptive threshold + sigmoid renorm (single thread) ----
    if (tid == 0) {
        float w[NEXP];
        float mx = -1e30f;
        #pragma unroll
        for (int e = 0; e < NEXP; ++e) { w[e] = s_logits[e]; mx = fmaxf(mx, w[e]); }
        float sum = 0.0f;
        #pragma unroll
        for (int e = 0; e < NEXP; ++e) { w[e] = __expf(w[e] - mx); sum += w[e]; }
        float inv = 1.0f / sum;
        #pragma unroll
        for (int e = 0; e < NEXP; ++e) w[e] *= inv;

        float sw[NEXP];
        #pragma unroll
        for (int e = 0; e < NEXP; ++e) sw[e] = w[e];
        #pragma unroll
        for (int i = 0; i < NEXP - 1; ++i)
            #pragma unroll
            for (int j = 0; j < NEXP - 1; ++j)
                if (sw[j] < sw[j + 1]) { float t = sw[j]; sw[j] = sw[j + 1]; sw[j + 1] = t; }

        float sw0 = sw[0];
        float max_component = 1.0f - sw0;
        float mean_w = 0.0f, entropy = 0.0f;
        #pragma unroll
        for (int e = 0; e < NEXP; ++e) {
            mean_w += w[e];
            entropy -= w[e] * __logf(w[e] + 1e-18f);
        }
        mean_w *= (1.0f / NEXP);
        float var = 0.0f;
        #pragma unroll
        for (int e = 0; e < NEXP; ++e) { float d = w[e] - mean_w; var += d * d; }
        float std_w = sqrtf(var * (1.0f / (NEXP - 1)));              // ddof=1
        float ent_c = 1.0f - entropy * (1.0f / 2.0794415416798357f); // /log(8)
        float mean_rest = (sw[1] + sw[2] + sw[3] + sw[4]) * 0.25f;   // k=min(4,7)
        float gap = (sw0 - mean_rest) / (sw0 + 1e-8f);
        gap = fminf(fmaxf(gap, 0.0f), 1.0f);

        float af = (*p_wmax) * max_component + (*p_went) * ent_c + (*p_wgap) * gap;
        float th = (*p_thr) * (0.5f + af);
        float min_th = fmaxf(0.05f, mean_w - 0.5f * std_w);
        float max_th = fminf(0.7f, sw0 - 0.1f * std_w);
        th = fminf(fmaxf(th, min_th), max_th);    // jnp.clip: max then min
        th = fminf(th, sw[1] * 0.9f);             // kth*0.9, MIN_EXPERTS_ACTIVE=2

        const int epoch = *p_epoch;
        float* orow = out + (size_t)n * NEXP;
        if (epoch < 20) {
            #pragma unroll
            for (int e = 0; e < NEXP; ++e) orow[e] = 1.0f / NEXP;
        } else {
            float tau = (epoch <= 25) ? fmaxf(0.1f, 1.0f - (epoch - 20) * 0.18f) : 0.1f;
            float s[NEXP], ssum = 0.0f;
            #pragma unroll
            for (int e = 0; e < NEXP; ++e) {
                s[e] = 1.0f / (1.0f + __expf(-(w[e] - th) / tau)); // hard+(soft-hard)
                ssum += s[e];
            }
            float rinv = 1.0f / fmaxf(ssum, 1e-8f);
            #pragma unroll
            for (int e = 0; e < NEXP; ++e) orow[e] = s[e] * rinv;
        }
    }
}

// ---------------------------------------------------------------------------
extern "C" void kernel_launch(void* const* d_in, const int* in_sizes, int n_in,
                              void* d_out, int out_size, void* d_ws, size_t ws_size,
                              hipStream_t stream) {
    const float* patch  = (const float*)d_in[0];
    const float* conv_w = (const float*)d_in[1];
    const float* conv_b = (const float*)d_in[2];
    const float* fc_w   = (const float*)d_in[3];
    const float* fc_b   = (const float*)d_in[4];
    const float* wmax   = (const float*)d_in[5];
    const float* went   = (const float*)d_in[6];
    const float* wgap   = (const float*)d_in[7];
    const float* thr    = (const float*)d_in[8];
    const int*   epoch  = (const int*)d_in[10];
    float* out = (float*)d_out;

    _Float16* wB = (_Float16*)d_ws;          // CPROJ*KTOT f16 = 168 KB, L2-resident

    const int N = in_sizes[0] / (CIN * HW);  // 16384 images

    const int wtot = CPROJ * KTOT;
    convert_weights_kernel<<<(wtot + 255) / 256, 256, 0, stream>>>(conv_w, wB);
    router_gate_kernel<<<N, 256, 0, stream>>>(patch, wB, conv_b, fc_w, fc_b,
                                              wmax, went, wgap, thr, epoch, out);
}
